// Mamba2Block_42125039239458
// MI455X (gfx1250) — matrix operand order repaired and checked by reading, not compiled, not run on hardware
//
#include <hip/hip_runtime.h>
#include <hip/hip_bf16.h>

// ---------------------------------------------------------------------------
// Mamba2 block for MI455X (gfx1250, wave32, WMMA + async LDS DMA).
//   1. addnorm:      res = h + r (-> d_out[1]); hs = rmsnorm(res) -> bf16
//   2. convert W1 (zero-padded to N=4416) / W2 to bf16
//   3. GEMM1 (WMMA bf16): proj[T,4416] = hs_bf16[T,1024] @ W1_bf16[1024,4416]
//   4. depthwise causal conv1d(K=4) + silu on xBC slice -> xbc[T,2304]
//   5. dt = softplus(dt_raw + bias), dA = exp(dt * -exp(A_log))
//   6. sequential SSM scan, 1 block per (b,head), state in registers
//   7. gated RMSNorm: y * silu(z), norm -> bf16
//   8. GEMM2 (WMMA bf16): out[T,1024] = y_bf16[T,2048] @ W2_bf16[2048,1024]
// GEMM A-tiles are staged with GLOBAL_LOAD_ASYNC_TO_LDS_B128 (ASYNCcnt):
// global -> LDS with zero staging VGPRs, overlapped with the WMMA block,
// closed by s_wait_asynccnt before the per-iteration barrier.
// ---------------------------------------------------------------------------

#define T_TOK   16384
#define SEQ     4096
#define BATCH   4
#define DMODEL  1024
#define INTER   2048
#define NHEADS  32
#define HEADDIM 64
#define DSTATE  128
#define CONVDIM 2304
#define DPROJ   4384
#define DPROJP  4416      // DPROJ padded to multiple of 64
#define EPS     1e-5f

typedef __bf16 bf16_t;
typedef __attribute__((ext_vector_type(16))) __bf16 v16bf;
typedef __attribute__((ext_vector_type(8)))  float  v8f;

__device__ __forceinline__ unsigned short f2bf(float f) {
    union { float f; unsigned u; } v; v.f = f;
    unsigned u = v.u;
    unsigned r = u + 0x7FFFu + ((u >> 16) & 1u);   // round to nearest even
    return (unsigned short)(r >> 16);
}

__device__ __forceinline__ float siluf(float x) {
    return x / (1.0f + __expf(-x));
}

// gfx1250 async global->LDS copy, 16B per lane, tracked by ASYNCcnt.
__device__ __forceinline__ void async_ld_b128(unsigned lds_off, const void* gptr) {
    asm volatile("global_load_async_to_lds_b128 %0, %1, off"
                 :: "v"(lds_off), "v"((unsigned long long)(uintptr_t)gptr)
                 : "memory");
}
__device__ __forceinline__ void wait_async0() {
    asm volatile("s_wait_asynccnt 0x0" ::: "memory");
}
// LDS byte offset of a __shared__ pointer (generic LDS address keeps the
// group-segment offset in the low 32 bits).
__device__ __forceinline__ unsigned lds_off(const void* p) {
    return (unsigned)(uintptr_t)p;
}

// ---------------------------------------------------------------------------
// 1. res = h + r ; hs = rmsnorm(res, w) -> bf16
// ---------------------------------------------------------------------------
__global__ __launch_bounds__(256) void addnorm_kernel(
    const float* __restrict__ h, const float* __restrict__ r,
    const float* __restrict__ w, float* __restrict__ resOut,
    unsigned short* __restrict__ hsb)
{
    const long token = blockIdx.x;
    const int  tid   = threadIdx.x;
    float v[4]; float ss = 0.f;
    #pragma unroll
    for (int it = 0; it < 4; ++it) {
        int i = tid + it * 256;
        float x = h[token * DMODEL + i] + r[token * DMODEL + i];
        v[it] = x;
        resOut[token * DMODEL + i] = x;
        ss += x * x;
    }
    #pragma unroll
    for (int off = 16; off; off >>= 1) ss += __shfl_xor(ss, off, 32);
    __shared__ float red[8];
    if ((tid & 31) == 0) red[tid >> 5] = ss;
    __syncthreads();
    float tot = red[0];
    #pragma unroll
    for (int wv = 1; wv < 8; ++wv) tot += red[wv];
    const float sc = rsqrtf(tot / (float)DMODEL + EPS);
    #pragma unroll
    for (int it = 0; it < 4; ++it) {
        int i = tid + it * 256;
        hsb[token * DMODEL + i] = f2bf(v[it] * sc * w[i]);
    }
}

// ---------------------------------------------------------------------------
// 2. f32 -> bf16 weight conversion with optional column zero-padding
// ---------------------------------------------------------------------------
__global__ __launch_bounds__(256) void cvt_pad_bf16_kernel(
    const float* __restrict__ in, unsigned short* __restrict__ out,
    int N, int Npad, long total)
{
    long i = (long)blockIdx.x * 256 + threadIdx.x;
    if (i >= total) return;
    int  n = (int)(i % Npad);
    long k = i / Npad;
    out[i] = (n < N) ? f2bf(in[k * N + n]) : (unsigned short)0;
}

// ---------------------------------------------------------------------------
// 3/8. WMMA bf16 GEMM: C[M,N] = A[M,K] @ B[K,N], f32 accumulate.
//      BM=256 BN=64 BK=32, 8 waves; wave w owns 2 M-subtiles x 4 N-tiles.
//      LDS double-buffered; A tiles staged by async global->LDS DMA.
//      Requires M % 256 == 0, N % 64 == 0, K % 32 == 0 (all guaranteed).
// ---------------------------------------------------------------------------
#define GBM 256
#define GBN 64
#define GBK 32
#define LSA 40   // LDS row stride (ushorts) for A rows of 32 K (+pad)
#define LSB 40   // LDS row stride (ushorts) for B^T rows of 32 K (+pad)

__global__ __launch_bounds__(256) void gemm_bf16_kernel(
    const unsigned short* __restrict__ A,   // M x K  (bf16 bits)
    const unsigned short* __restrict__ B,   // K x N  (bf16 bits)
    float* __restrict__ C,                  // M x N
    int M, int N, int K)
{
    __shared__ __align__(16) unsigned short lA[2][GBM * LSA];
    __shared__ __align__(16) unsigned short lB[2][GBN * LSB];

    const int tid  = threadIdx.x;
    const int lane = tid & 31;
    const int wid  = tid >> 5;
    const int half = lane >> 4;
    const int l16  = lane & 15;

    const int n0 = blockIdx.x * GBN;
    const int m0 = blockIdx.y * GBM;

    // per-thread staging coordinates (32-bit offsets everywhere; < 2^31)
    int arow[4], aq[4];
    #pragma unroll
    for (int it = 0; it < 4; ++it) {
        int j = tid + it * 256;          // 0..1023 -> 256 rows x 4 quads
        arow[it] = j >> 2;
        aq[it]   = j & 3;
    }
    const int kb = tid >> 3;             // B: K row within tile (0..31)
    const int nb = (tid & 7) * 8;        // B: 8 consecutive N elements

    v8f acc[2][4];
    #pragma unroll
    for (int mi = 0; mi < 2; ++mi)
        #pragma unroll
        for (int c = 0; c < 4; ++c)
            #pragma unroll
            for (int e = 0; e < 8; ++e) acc[mi][c][e] = 0.f;

    uint4 tb;                            // only B is staged through VGPRs

    // --- prologue: tile 0 -> lds[0] (A async, B via regs) ---
    #pragma unroll
    for (int it = 0; it < 4; ++it)
        async_ld_b128(lds_off(&lA[0][arow[it] * LSA + aq[it] * 8]),
                      A + (unsigned)((m0 + arow[it]) * K + aq[it] * 8));
    tb = *(const uint4*)(B + (unsigned)(kb * N + n0 + nb));
    {
        lB[0][(nb + 0) * LSB + kb] = (unsigned short)(tb.x);
        lB[0][(nb + 1) * LSB + kb] = (unsigned short)(tb.x >> 16);
        lB[0][(nb + 2) * LSB + kb] = (unsigned short)(tb.y);
        lB[0][(nb + 3) * LSB + kb] = (unsigned short)(tb.y >> 16);
        lB[0][(nb + 4) * LSB + kb] = (unsigned short)(tb.z);
        lB[0][(nb + 5) * LSB + kb] = (unsigned short)(tb.z >> 16);
        lB[0][(nb + 6) * LSB + kb] = (unsigned short)(tb.w);
        lB[0][(nb + 7) * LSB + kb] = (unsigned short)(tb.w >> 16);
    }
    wait_async0();
    __syncthreads();

    int cur = 0;
    for (int kk = 0; kk < K; kk += GBK) {
        const bool more = (kk + GBK) < K;

        // --- issue next tile: A async DMA into lds[cur^1], B into regs ---
        if (more) {
            int kn = kk + GBK;
            #pragma unroll
            for (int it = 0; it < 4; ++it)
                async_ld_b128(lds_off(&lA[cur ^ 1][arow[it] * LSA + aq[it] * 8]),
                              A + (unsigned)((m0 + arow[it]) * K + kn + aq[it] * 8));
            tb = *(const uint4*)(B + (unsigned)((kn + kb) * N + n0 + nb));
            if (kk + 2 * GBK < K)   // stream 2 tiles ahead into L2
                __builtin_prefetch(A + (unsigned)((m0 + arow[0]) * K + kk + 2 * GBK), 0, 0);
        }

        // --- A fragments from lA[cur]: row = mi*128 + 16*wid + l16 ---
        union { v16bf v; uint4 q[2]; } af[2];
        #pragma unroll
        for (int mi = 0; mi < 2; ++mi) {
            const unsigned short* ap = &lA[cur][(mi * 128 + wid * 16 + l16) * LSA + half * 8];
            af[mi].q[0] = *(const uint4*)ap;
            af[mi].q[1] = *(const uint4*)(ap + 16);
        }

        #pragma unroll
        for (int c = 0; c < 4; ++c) {
            // B fragment: col = 16*c + l16; K = half*8 + (0..7), 16 + half*8 + (0..7)
            union { v16bf v; uint4 q[2]; } bfu;
            const unsigned short* bp = &lB[cur][(c * 16 + l16) * LSB + half * 8];
            bfu.q[0] = *(const uint4*)bp;
            bfu.q[1] = *(const uint4*)(bp + 16);
            #pragma unroll
            for (int mi = 0; mi < 2; ++mi)
                acc[mi][c] = __builtin_amdgcn_wmma_f32_16x16x32_bf16(
                    false, af[mi].v, false, bfu.v, (short)0, acc[mi][c], false, false);
        }

        // --- commit B regs, then close the async pipe before the barrier ---
        if (more) {
            lB[cur ^ 1][(nb + 0) * LSB + kb] = (unsigned short)(tb.x);
            lB[cur ^ 1][(nb + 1) * LSB + kb] = (unsigned short)(tb.x >> 16);
            lB[cur ^ 1][(nb + 2) * LSB + kb] = (unsigned short)(tb.y);
            lB[cur ^ 1][(nb + 3) * LSB + kb] = (unsigned short)(tb.y >> 16);
            lB[cur ^ 1][(nb + 4) * LSB + kb] = (unsigned short)(tb.z);
            lB[cur ^ 1][(nb + 5) * LSB + kb] = (unsigned short)(tb.z >> 16);
            lB[cur ^ 1][(nb + 6) * LSB + kb] = (unsigned short)(tb.w);
            lB[cur ^ 1][(nb + 7) * LSB + kb] = (unsigned short)(tb.w >> 16);
            wait_async0();
        }
        __syncthreads();
        cur ^= 1;
    }

    // --- store: elem r -> (M = m0 + mi*128 + 16*wid + r + 8*half, N = n0 + 16*c + l16)
    #pragma unroll
    for (int mi = 0; mi < 2; ++mi) {
        #pragma unroll
        for (int c = 0; c < 4; ++c) {
            int n = n0 + c * 16 + l16;
            #pragma unroll
            for (int r = 0; r < 8; ++r) {
                int m = m0 + mi * 128 + wid * 16 + r + half * 8;
                C[(unsigned)(m * N + n)] = acc[mi][c][r];
            }
        }
    }
}

// ---------------------------------------------------------------------------
// 4. depthwise causal conv1d (K=4) + silu on proj[..., 2048:2048+2304]
// ---------------------------------------------------------------------------
__global__ __launch_bounds__(256) void conv_silu_kernel(
    const float* __restrict__ proj, const float* __restrict__ cw,
    const float* __restrict__ cb, float* __restrict__ xbc)
{
    long idx = (long)blockIdx.x * 256 + threadIdx.x;
    if (idx >= (long)T_TOK * CONVDIM) return;
    int  c     = (int)(idx % CONVDIM);
    long token = idx / CONVDIM;
    int  s     = (int)(token % SEQ);
    float acc = cb[c];
    #pragma unroll
    for (int k = 0; k < 4; ++k) {
        int sp = s - 3 + k;
        if (sp >= 0)
            acc += proj[(token - (3 - k)) * DPROJP + INTER + c] * cw[c * 4 + k];
    }
    xbc[idx] = siluf(acc);
}

// ---------------------------------------------------------------------------
// 5. dt = softplus(dt_raw + bias); dA = exp(dt * -exp(A_log))
// ---------------------------------------------------------------------------
__global__ __launch_bounds__(256) void dtda_kernel(
    const float* __restrict__ proj, const float* __restrict__ dt_bias,
    const float* __restrict__ A_log, float* __restrict__ dtb,
    float* __restrict__ dab)
{
    long idx = (long)blockIdx.x * 256 + threadIdx.x;
    if (idx >= (long)T_TOK * NHEADS) return;
    int  hh    = (int)(idx & (NHEADS - 1));
    long token = idx >> 5;
    float x  = proj[token * DPROJP + (INTER + CONVDIM) + hh] + dt_bias[hh];
    float dt = (x > 20.f) ? x : log1pf(__expf(x));
    float Av = -__expf(A_log[hh]);
    dtb[idx] = dt;
    dab[idx] = __expf(dt * Av);
}

// ---------------------------------------------------------------------------
// 6. SSM scan. One block per (b, head). 8 waves x 8 p-values, lane -> 4 n.
//    state[p][n] kept in registers (32 f32 per lane).
// ---------------------------------------------------------------------------
__global__ __launch_bounds__(256) void scan_kernel(
    const float* __restrict__ xbc, const float* __restrict__ dtb,
    const float* __restrict__ dab, const float* __restrict__ Dp,
    float* __restrict__ y)
{
    const int bh   = blockIdx.x;
    const int b    = bh >> 5;
    const int hh   = bh & 31;
    const int lane = threadIdx.x & 31;
    const int wv   = threadIdx.x >> 5;
    const int pbase = hh * HEADDIM + wv * 8;
    const int nbase = lane * 4;
    const float Dh = Dp[hh];

    float st[8][4];
    #pragma unroll
    for (int i = 0; i < 8; ++i)
        #pragma unroll
        for (int j = 0; j < 4; ++j) st[i][j] = 0.f;

    for (int t = 0; t < SEQ; ++t) {
        const long   token = (long)b * SEQ + t;
        const float* row   = xbc + token * CONVDIM;
        const float  dAv   = dab[token * NHEADS + hh];
        const float  dtv   = dtb[token * NHEADS + hh];
        float Bv[4], Cv[4], xv[8];
        #pragma unroll
        for (int j = 0; j < 4; ++j) {
            Bv[j] = row[INTER + nbase + j];
            Cv[j] = row[INTER + DSTATE + nbase + j];
        }
        #pragma unroll
        for (int i = 0; i < 8; ++i) xv[i] = row[pbase + i];

        float yp[8];
        #pragma unroll
        for (int i = 0; i < 8; ++i) {
            float dx = dtv * xv[i];
            float s  = 0.f;
            #pragma unroll
            for (int j = 0; j < 4; ++j) {
                st[i][j] = st[i][j] * dAv + dx * Bv[j];
                s += st[i][j] * Cv[j];
            }
            yp[i] = s;
        }
        #pragma unroll
        for (int i = 0; i < 8; ++i) {
            #pragma unroll
            for (int off = 16; off; off >>= 1)
                yp[i] += __shfl_xor(yp[i], off, 32);
        }
        if (lane == 0) {
            #pragma unroll
            for (int i = 0; i < 8; ++i)
                y[token * INTER + pbase + i] = yp[i] + Dh * xv[i];
        }
    }
}

// ---------------------------------------------------------------------------
// 7. gated RMSNorm: g = y * silu(z); out_bf16 = rmsnorm(g) * w
// ---------------------------------------------------------------------------
__global__ __launch_bounds__(256) void gatenorm_kernel(
    const float* __restrict__ y, const float* __restrict__ proj,
    const float* __restrict__ w, unsigned short* __restrict__ yb)
{
    const long token = blockIdx.x;
    const int  tid   = threadIdx.x;
    float g[8]; float ss = 0.f;
    #pragma unroll
    for (int it = 0; it < 8; ++it) {
        int i = tid + it * 256;
        float zv = proj[token * DPROJP + i];
        float gv = y[token * INTER + i] * siluf(zv);
        g[it] = gv;
        ss += gv * gv;
    }
    #pragma unroll
    for (int off = 16; off; off >>= 1) ss += __shfl_xor(ss, off, 32);
    __shared__ float red[8];
    if ((tid & 31) == 0) red[tid >> 5] = ss;
    __syncthreads();
    float tot = red[0];
    #pragma unroll
    for (int wv = 1; wv < 8; ++wv) tot += red[wv];
    const float sc = rsqrtf(tot / (float)INTER + EPS);
    #pragma unroll
    for (int it = 0; it < 8; ++it) {
        int i = tid + it * 256;
        yb[token * INTER + i] = f2bf(g[it] * sc * w[i]);
    }
}

// ---------------------------------------------------------------------------
// launcher
// ---------------------------------------------------------------------------
extern "C" void kernel_launch(void* const* d_in, const int* in_sizes, int n_in,
                              void* d_out, int out_size, void* d_ws, size_t ws_size,
                              hipStream_t stream) {
    const float* h      = (const float*)d_in[0];
    const float* r      = (const float*)d_in[1];
    const float* normw  = (const float*)d_in[2];
    const float* inw    = (const float*)d_in[3];
    const float* convw  = (const float*)d_in[4];
    const float* convb  = (const float*)d_in[5];
    const float* dtbias = (const float*)d_in[6];
    const float* alog   = (const float*)d_in[7];
    const float* Dp     = (const float*)d_in[8];
    const float* gatew  = (const float*)d_in[9];
    const float* outw   = (const float*)d_in[10];

    float* out    = (float*)d_out;                        // [T, 1024]
    float* resOut = out + (size_t)T_TOK * DMODEL;         // [T, 1024]

    // ---- workspace carve-up (bf16 section first, 16B-aligned throughout) ----
    char* ws = (char*)d_ws;
    unsigned short* hs_bf = (unsigned short*)ws;                       // T*1024
    unsigned short* w1_bf = hs_bf + (size_t)T_TOK * DMODEL;            // 1024*4416 (padded)
    unsigned short* w2_bf = w1_bf + (size_t)DMODEL * DPROJP;           // 2048*1024
    unsigned short* y_bf  = w2_bf + (size_t)INTER * DMODEL;            // T*2048
    size_t ush = (size_t)T_TOK * DMODEL + (size_t)DMODEL * DPROJP +
                 (size_t)INTER * DMODEL + (size_t)T_TOK * INTER;
    float* proj = (float*)(ws + ((ush * 2 + 255) & ~(size_t)255));     // T*4416 (padded)
    float* xbc  = proj + (size_t)T_TOK * DPROJP;                       // T*2304
    float* dtb  = xbc  + (size_t)T_TOK * CONVDIM;                      // T*32
    float* dab  = dtb  + (size_t)T_TOK * NHEADS;                       // T*32
    float* ybuf = dab  + (size_t)T_TOK * NHEADS;                       // T*2048

    // 1. residual add + rmsnorm -> bf16
    addnorm_kernel<<<T_TOK, 256, 0, stream>>>(h, r, normw, resOut, hs_bf);

    // 2. weight conversions (W1 zero-padded to 4416 columns)
    {
        long n1 = (long)DMODEL * DPROJP;
        cvt_pad_bf16_kernel<<<(unsigned)((n1 + 255) / 256), 256, 0, stream>>>(
            inw, w1_bf, DPROJ, DPROJP, n1);
        long n2 = (long)INTER * DMODEL;
        cvt_pad_bf16_kernel<<<(unsigned)((n2 + 255) / 256), 256, 0, stream>>>(
            outw, w2_bf, DMODEL, DMODEL, n2);
    }

    // 3. GEMM1: proj = hs @ W1   (M=16384, N=4416, K=1024)
    {
        dim3 grid(DPROJP / GBN, T_TOK / GBM);
        gemm_bf16_kernel<<<grid, 256, 0, stream>>>(hs_bf, w1_bf, proj,
                                                   T_TOK, DPROJP, DMODEL);
    }

    // 4. conv1d + silu
    {
        long n = (long)T_TOK * CONVDIM;
        conv_silu_kernel<<<(unsigned)((n + 255) / 256), 256, 0, stream>>>(proj, convw, convb, xbc);
    }

    // 5. dt / dA
    {
        long n = (long)T_TOK * NHEADS;
        dtda_kernel<<<(unsigned)((n + 255) / 256), 256, 0, stream>>>(proj, dtbias, alog, dtb, dab);
    }

    // 6. scan
    scan_kernel<<<BATCH * NHEADS, 256, 0, stream>>>(xbc, dtb, dab, Dp, ybuf);

    // 7. gated rmsnorm -> bf16
    gatenorm_kernel<<<T_TOK, 256, 0, stream>>>(ybuf, proj, gatew, y_bf);

    // 8. GEMM2: out = y @ W2   (M=16384, N=1024, K=2048)
    {
        dim3 grid(DMODEL / GBN, T_TOK / GBM);
        gemm_bf16_kernel<<<grid, 256, 0, stream>>>(y_bf, w2_bf, out,
                                                   T_TOK, DMODEL, INTER);
    }
}
